// KNNModule_31301721653637
// MI455X (gfx1250) — matrix operand order repaired and checked by reading, not compiled, not run on hardware
//
#include <hip/hip_runtime.h>

// ---------------------------------------------------------------------------
// Batched kNN for MI455X (gfx1250, wave32, WMMA).
//   d2[i][j] = |xi|^2 + |xj|^2 - 2 xi.xj ; mask cross-batch & self ; top-32.
// Gram tiles via split-bf16 WMMA (hi*hi + hi*lo + lo*hi, f32 accumulate):
// near-fp32 precision on the codegen-confirmed v_wmma_f32_16x16x32_bf16 path.
// batch[] is sorted -> contiguous segments -> whole-tile cross-batch reject.
// ---------------------------------------------------------------------------

typedef __bf16 bf16_t;
typedef __attribute__((ext_vector_type(8)))  bf16_t v8bf;
typedef __attribute__((ext_vector_type(16))) bf16_t v16bf;
typedef __attribute__((ext_vector_type(8)))  float  v8f;

#define KNN_K   32
#define BIGF    1e30f
#define TILE    16
#define WAVES   8
#define BLOCK   (WAVES * 32)

// ---------------------------------------------------------------------------
// Prep: x(f32) -> xhi(bf16), xlo(bf16 = x - xhi), sq = rowwise |x|^2.
// One workgroup per row, blockDim.x == D (256).
// ---------------------------------------------------------------------------
__global__ void knn_prep(const float* __restrict__ x,
                         bf16_t* __restrict__ xhi,
                         bf16_t* __restrict__ xlo,
                         float*  __restrict__ sq,
                         int D) {
  const int row = blockIdx.x;
  const int t   = threadIdx.x;
  const size_t off = (size_t)row * D + t;
  float v = x[off];
  bf16_t h = (bf16_t)v;
  bf16_t l = (bf16_t)(v - (float)h);
  xhi[off] = h;
  xlo[off] = l;

  __shared__ float red[256];
  red[t] = v * v;
  __syncthreads();
  for (int s = 128; s > 0; s >>= 1) {
    if (t < s) red[t] += red[t + s];
    __syncthreads();
  }
  if (t == 0) sq[row] = red[0];
}

// ---------------------------------------------------------------------------
// Main: one 16-row tile per workgroup; 8 waves split the column tiles.
// Per 16x16 tile: 24 WMMAs (8 K-chunks x {hh, hl, lh}), mask, stage to LDS,
// lanes 0..15 maintain per-wave sorted top-32 per row; final 8-way merge.
//
// Fragment layouts used by this code (wave32):
//   A 16x32 bf16 : lane L -> row M=L&15 ; elems 0..15 = K (16*(L>>4))+0..15
//   B 32x16 bf16 : lane L -> col N=L&15 ; elems 0..15 = K (16*(L>>4))+0..15
//   C/D 16x16 f32: lane L -> col N=L&15 ; VGPR v -> row M = v + 8*(L>>4)
// ---------------------------------------------------------------------------
__global__ __launch_bounds__(BLOCK) void knn_main(
    const bf16_t* __restrict__ xhi, const bf16_t* __restrict__ xlo,
    const float* __restrict__ sq, const int* __restrict__ batch,
    float* __restrict__ outD, float* __restrict__ outI,
    int N, int D) {

  __shared__ float s_rowSq[TILE];
  __shared__ int   s_rowB[TILE];
  __shared__ float s_stage[WAVES][TILE][TILE];
  __shared__ float s_topD[WAVES][TILE][KNN_K];
  __shared__ int   s_topI[WAVES][TILE][KNN_K];
  __shared__ float s_fD[TILE][KNN_K];
  __shared__ int   s_fI[TILE][KNN_K];

  const int tid  = threadIdx.x;
  const int wave = tid >> 5;
  const int lane = tid & 31;
  const int r0   = blockIdx.x * TILE;

  if (tid < TILE) {
    s_rowSq[tid] = sq[r0 + tid];
    s_rowB[tid]  = batch[r0 + tid];
  }
  for (int i = tid; i < WAVES * TILE * KNN_K; i += BLOCK) {
    (&s_topD[0][0][0])[i] = BIGF;
    (&s_topI[0][0][0])[i] = -1;
  }
  __syncthreads();

  const int rbMin = s_rowB[0];          // batch sorted -> tile batch range
  const int rbMax = s_rowB[TILE - 1];

  const int hi    = lane >> 4;          // half-wave select
  const int lm    = lane & 15;
  const int aBase = hi * 16;            // A fragment K sub-offset
  const int bBase = hi * 16;            // B fragment K sub-offset
  const int nCT   = N / TILE;
  const int nKC   = D / 32;             // K chunks of 32 (D=256 -> 8)

  const size_t aOff = (size_t)(r0 + lm) * D;   // A row this lane services

  for (int ct = wave; ct < nCT; ct += WAVES) {
    const int c0 = ct * TILE;
    // Whole-tile cross-batch reject (batch sorted => segment ranges).
    if (batch[c0 + TILE - 1] < rbMin || batch[c0] > rbMax) continue;

    // Prefetch next column tile's B rows into cache (global_prefetch_b8).
    if (ct + WAVES < nCT)
      __builtin_prefetch(xhi + (size_t)((ct + WAVES) * TILE + lm) * D, 0, 0);

    const int    bCol  = c0 + lm;
    const size_t bOff  = (size_t)bCol * D;
    const float  colSq = sq[bCol];
    const int    colB  = batch[bCol];

    v8f acc = {};
    #pragma unroll
    for (int kc = 0; kc < 8; ++kc) {    // D == 256 assumed (nKC == 8)
      const int k0 = kc * 32;
      (void)nKC;
      // A fragments: two 16B chunks per lane, fused into v16bf.
      v8bf ah0 = *(const v8bf*)(xhi + aOff + k0 + aBase);
      v8bf ah1 = *(const v8bf*)(xhi + aOff + k0 + 8 + aBase);
      v8bf al0 = *(const v8bf*)(xlo + aOff + k0 + aBase);
      v8bf al1 = *(const v8bf*)(xlo + aOff + k0 + 8 + aBase);
      v16bf ah = __builtin_shufflevector(ah0, ah1,
          0,1,2,3,4,5,6,7,8,9,10,11,12,13,14,15);
      v16bf al = __builtin_shufflevector(al0, al1,
          0,1,2,3,4,5,6,7,8,9,10,11,12,13,14,15);
      // B fragments: one contiguous 32B chunk per lane.
      v16bf bh = *(const v16bf*)(xhi + bOff + k0 + bBase);
      v16bf bl = *(const v16bf*)(xlo + bOff + k0 + bBase);
      // Split-bf16 product: hi*hi + hi*lo + lo*hi  (~fp32-quality dot).
      acc = __builtin_amdgcn_wmma_f32_16x16x32_bf16(false, ah, false, bh,
                                                    (short)0, acc, false, false);
      acc = __builtin_amdgcn_wmma_f32_16x16x32_bf16(false, ah, false, bl,
                                                    (short)0, acc, false, false);
      acc = __builtin_amdgcn_wmma_f32_16x16x32_bf16(false, al, false, bh,
                                                    (short)0, acc, false, false);
    }

    // Mask + stage distances (wave-private LDS region: same-wave DS ops are
    // in-order, no barrier needed between the stores and the scan below).
    #pragma unroll
    for (int v = 0; v < 8; ++v) {
      const int rrel = v + 8 * hi;
      const int row  = r0 + rrel;
      float d2 = s_rowSq[rrel] + colSq - 2.0f * acc[v];
      const bool valid = (s_rowB[rrel] == colB) && (row != bCol);
      s_stage[wave][rrel][lm] = valid ? d2 : BIGF;
    }

    // Streaming insertion: lane r owns row r; per-wave sorted top-32 in LDS.
    if (lane < TILE) {
      float* td = &s_topD[wave][lane][0];
      int*   ti = &s_topI[wave][lane][0];
      #pragma unroll 1
      for (int j = 0; j < TILE; ++j) {
        const float d = s_stage[wave][lane][j];
        if (d < td[KNN_K - 1]) {
          int p = KNN_K - 1;
          while (p > 0 && td[p - 1] > d) {
            td[p] = td[p - 1]; ti[p] = ti[p - 1]; --p;
          }
          td[p] = d; ti[p] = c0 + j;
        }
      }
    }
  }
  __syncthreads();

  // Merge 8 sorted per-wave lists per row; threads 0..15, one row each.
  if (tid < TILE) {
    float* fd = &s_fD[tid][0];
    int*   fi = &s_fI[tid][0];
    for (int k = 0; k < KNN_K; ++k) { fd[k] = BIGF; fi[k] = -1; }
    for (int w = 0; w < WAVES; ++w) {
      for (int k = 0; k < KNN_K; ++k) {
        const float d = s_topD[w][tid][k];
        if (d >= fd[KNN_K - 1]) break;     // source list sorted ascending
        const int c = s_topI[w][tid][k];
        int p = KNN_K - 1;
        while (p > 0 && fd[p - 1] > d) {
          fd[p] = fd[p - 1]; fi[p] = fi[p - 1]; --p;
        }
        fd[p] = d; fi[p] = c;
      }
    }
    const size_t base = (size_t)(r0 + tid) * KNN_K;
    for (int k = 0; k < KNN_K; ++k) {
      outD[base + k] = fd[k];
      outI[base + k] = (float)fi[k];      // index output, float-encoded
    }
  }
}

// ---------------------------------------------------------------------------
extern "C" void kernel_launch(void* const* d_in, const int* in_sizes, int n_in,
                              void* d_out, int out_size, void* d_ws, size_t ws_size,
                              hipStream_t stream) {
  const float* x     = (const float*)d_in[0];
  const int*   batch = (const int*)d_in[1];
  const int N = in_sizes[1];
  const int D = in_sizes[0] / N;   // 8192 x 256

  // Workspace: xhi | xlo | sq  (~8.4 MB for N=8192, D=256)
  bf16_t* xhi = (bf16_t*)d_ws;
  bf16_t* xlo = xhi + (size_t)N * D;
  float*  sq  = (float*)(xlo + (size_t)N * D);

  float* outD = (float*)d_out;
  float* outI = outD + (size_t)N * KNN_K;

  knn_prep<<<N, D, 0, stream>>>(x, xhi, xlo, sq, D);
  knn_main<<<N / TILE, BLOCK, 0, stream>>>(xhi, xlo, sq, batch,
                                           outD, outI, N, D);
}
